// VarGATEncoder_38920993636578
// MI455X (gfx1250) — matrix operand order repaired and checked by reading, not compiled, not run on hardware
//
#include <hip/hip_runtime.h>
#include <hip/hip_bf16.h>

typedef __attribute__((ext_vector_type(16))) __bf16       v16bf;
typedef __attribute__((ext_vector_type(8)))  float        v8f;
typedef __attribute__((ext_vector_type(4)))  unsigned int u32x4;
typedef __attribute__((ext_vector_type(8)))  int          i32x8;
typedef __attribute__((ext_vector_type(4)))  int          i32x4;

#define XSTRIDE 132   // f32 elems per LDS row (512B data + 16B TDM pad), 16B aligned
#define WSTRIDE 136   // bf16 elems per LDS row, 16B aligned

// ---------------------------------------------------------------------------
// Y[M,OC] = X[M,128] @ W[128,OC]; fp32 in/out.
// X tile staged by the Tensor Data Mover (raw f32, HW zero-fill for tail rows,
// HW row padding for bank spread); W staged manually transposed as split-bf16.
// Split-bf16 product: ah*bh + ah*bl + al*bh  (~fp32 accuracy), fp32 WMMA accum.
// grid.x = ceil(M/128), block = 256 (8 waves).
// ---------------------------------------------------------------------------
template<int OC>
__global__ __launch_bounds__(256)
void wmma_gemm_k128(const float* __restrict__ X, const float* __restrict__ W,
                    float* __restrict__ Y, int M)
{
    __shared__ float  sXf[128 * XSTRIDE];   // filled by TDM
    __shared__ __bf16 sWh[OC * WSTRIDE];    // transposed: [col][k], hi part
    __shared__ __bf16 sWl[OC * WSTRIDE];    // transposed: [col][k], lo part

    const int tid  = threadIdx.x;
    const int row0 = blockIdx.x * 128;

    // ---- wave 0: issue TDM load of the 128x128 f32 X tile into sXf ----
    if (tid < 32) {
        const unsigned long long ga =
            (unsigned long long)(uintptr_t)(X + (size_t)row0 * 128);
        const unsigned ldsb = (unsigned)(unsigned long long)(uintptr_t)(void*)&sXf[0];
        const unsigned rows = (unsigned)(M - row0);   // tensor_dim1: HW zero-fills OOB rows
        u32x4 g0 = { 1u,                                            // count=1
                     ldsb,                                          // lds_addr
                     (unsigned)(ga & 0xFFFFFFFFull),                // global_addr lo
                     (unsigned)((ga >> 32) & 0x01FFFFFFull) | (2u << 30) }; // addr hi | type=2
        i32x8 g1 = { (int)((2u << 16) | (1u << 20) | (6u << 22) | (3u << 25)),
                     //      data=4B     pad_en      ivl=128dw     pad=4dw
                     (int)(128u << 16),                             // tensor_dim0 = 128
                     (int)((rows & 0xFFFFu) << 16),                 // dim0 hi | tensor_dim1 lo
                     (int)(((rows >> 16) & 0xFFFFu) | (128u << 16)),// dim1 hi | tile_dim0=128
                     128,                                           // tile_dim1=128, tile_dim2=0
                     128,                                           // tensor_dim0_stride = 128
                     0, 0 };
        i32x4 gz = { 0, 0, 0, 0 };
#if defined(__clang_major__) && __clang_major__ >= 23
        i32x8 gz8 = { 0, 0, 0, 0, 0, 0, 0, 0 };
        __builtin_amdgcn_tensor_load_to_lds(g0, g1, gz, gz, gz8, 0);
#else
        __builtin_amdgcn_tensor_load_to_lds(g0, g1, gz, gz, 0);
#endif
        __builtin_amdgcn_s_wait_tensorcnt(0);
    }

    // ---- all waves: stage W transposed, split into bf16 hi/lo ----
    for (int j = tid; j < 128 * OC; j += 256) {
        const int k = j / OC, c = j % OC;
        const float  w  = W[j];
        const __bf16 wh = (__bf16)w;
        sWh[c * WSTRIDE + k] = wh;
        sWl[c * WSTRIDE + k] = (__bf16)(w - (float)wh);
    }
    __syncthreads();

    const int wave = tid >> 5;
    const int lane = tid & 31;
    const int m0   = wave * 16;
    const int lm   = lane & 15;
    const int lk8  = (lane >> 4) * 8;    // unused; A and B both use lk16
    const int lk16 = (lane >> 4) * 16;   // B-matrix K sub-offset

    constexpr int NT = OC / 16;
    v8f acc[NT];
#pragma unroll
    for (int t = 0; t < NT; ++t) acc[t] = (v8f){0,0,0,0,0,0,0,0};

#pragma unroll
    for (int kb = 0; kb < 128; kb += 32) {
        // A fragment from f32 LDS, split hi/lo at build time
        const float* xr = &sXf[(m0 + lm) * XSTRIDE + kb];
        float av[16];
#pragma unroll
        for (int i = 0; i < 8; ++i) av[i]     = xr[lk16 + i];
#pragma unroll
        for (int i = 0; i < 8; ++i) av[8 + i] = xr[lk16 + 8 + i];
        v16bf ah, al;
#pragma unroll
        for (int i = 0; i < 16; ++i) {
            const __bf16 h = (__bf16)av[i];
            ah[i] = h;
            al[i] = (__bf16)(av[i] - (float)h);
        }

#pragma unroll
        for (int t = 0; t < NT; ++t) {
            const __bf16* bhp = &sWh[(t * 16 + lm) * WSTRIDE + kb + lk16];
            const __bf16* blp = &sWl[(t * 16 + lm) * WSTRIDE + kb + lk16];
            v16bf bh, bl;
#pragma unroll
            for (int i = 0; i < 16; ++i) { bh[i] = bhp[i]; bl[i] = blp[i]; }
            acc[t] = __builtin_amdgcn_wmma_f32_16x16x32_bf16(
                false, ah, false, bh, (short)0, acc[t], false, false);
            acc[t] = __builtin_amdgcn_wmma_f32_16x16x32_bf16(
                false, ah, false, bl, (short)0, acc[t], false, false);
            acc[t] = __builtin_amdgcn_wmma_f32_16x16x32_bf16(
                false, al, false, bh, (short)0, acc[t], false, false);
        }
    }

    // C/D layout: lane -> N = lane&15 ; element j -> M = (lane>>4)*8 + j
    const int om = (lane >> 4) * 8;
    const int on = lane & 15;
#pragma unroll
    for (int t = 0; t < NT; ++t) {
#pragma unroll
        for (int j = 0; j < 8; ++j) {
            const int gr = row0 + m0 + om + j;
            if (gr < M) Y[(size_t)gr * OC + t * 16 + on] = acc[t][j];
        }
    }
}

// ---------------------------------------------------------------------------
// per-node attention logits: als[n] = h[n]·a_src, ald[n] = h[n]·a_dst
// ---------------------------------------------------------------------------
__global__ void node_dots_kernel(const float* __restrict__ h,
                                 const float* __restrict__ a_s,
                                 const float* __restrict__ a_d,
                                 float* __restrict__ als, float* __restrict__ ald,
                                 int n, int C)
{
    const int node = (int)((blockIdx.x * (long long)blockDim.x + threadIdx.x) >> 5);
    const int lane = threadIdx.x & 31;
    if (node >= n) return;
    float s = 0.f, d = 0.f;
    for (int c = lane; c < C; c += 32) {
        const float v = h[(size_t)node * C + c];
        s += v * a_s[c];
        d += v * a_d[c];
    }
#pragma unroll
    for (int off = 16; off > 0; off >>= 1) {
        s += __shfl_down(s, off, 32);
        d += __shfl_down(d, off, 32);
    }
    if (lane == 0) { als[node] = s; ald[node] = d; }
}

__global__ void fill_u32_kernel(unsigned int* __restrict__ p, unsigned int v, long long count)
{
    const long long i = blockIdx.x * (long long)blockDim.x + threadIdx.x;
    if (i < count) p[i] = v;
}

__device__ __forceinline__ void edge_endpoints(const int* src, const int* dst,
                                               int E_real, int e, int& s, int& d)
{
    if (e < E_real) { s = src[e]; d = dst[e]; }
    else            { s = d = e - E_real; }      // appended self loops
}

// pass 1: segment max of leaky_relu(als[src]+ald[dst], 0.2) over dst
__global__ void edge_max_kernel(const int* __restrict__ src, const int* __restrict__ dst,
                                int E_real, int nloop,
                                const float* __restrict__ als, const float* __restrict__ ald,
                                unsigned int* __restrict__ emax_bits)
{
    const int e = blockIdx.x * blockDim.x + threadIdx.x;
    if (e >= E_real + nloop) return;
    int s, d; edge_endpoints(src, dst, E_real, e, s, d);
    float v = als[s] + ald[d];
    v = v > 0.f ? v : 0.2f * v;
    if (v >= 0.f) atomicMax((int*)&emax_bits[d], __float_as_int(v));
    else          atomicMin(&emax_bits[d], __float_as_uint(v));
}

// pass 2: ee = exp(e - emax[dst]); den[dst] += ee
__global__ void edge_exp_kernel(const int* __restrict__ src, const int* __restrict__ dst,
                                int E_real, int nloop,
                                const float* __restrict__ als, const float* __restrict__ ald,
                                const unsigned int* __restrict__ emax_bits,
                                float* __restrict__ den, float* __restrict__ ee)
{
    const int e = blockIdx.x * blockDim.x + threadIdx.x;
    if (e >= E_real + nloop) return;
    int s, d; edge_endpoints(src, dst, E_real, e, s, d);
    float v = als[s] + ald[d];
    v = v > 0.f ? v : 0.2f * v;
    const float m  = __uint_as_float(emax_bits[d]);
    const float ev = __expf(v - m);
    ee[e] = ev;
    atomicAdd(&den[d], ev);
}

// pass 3: agg[dst] += h[src] * (ee/den[dst]); one wave32 per edge, float4 rows
__global__ void edge_scatter_kernel(const int* __restrict__ src, const int* __restrict__ dst,
                                    int E_real, int nloop,
                                    const float* __restrict__ ee, const float* __restrict__ den,
                                    const float* __restrict__ h, float* __restrict__ agg, int C)
{
    const int e    = (int)((blockIdx.x * (long long)blockDim.x + threadIdx.x) >> 5);
    const int lane = threadIdx.x & 31;
    if (e >= E_real + nloop) return;
    int s, d; edge_endpoints(src, dst, E_real, e, s, d);
    const float alpha = ee[e] / (den[d] + 1e-16f);
    if (lane < (C >> 2)) {
        const float4 v = *(const float4*)(h + (size_t)s * C + lane * 4);
        float* base = agg + (size_t)d * C + lane * 4;
        atomicAdd(base + 0, v.x * alpha);
        atomicAdd(base + 1, v.y * alpha);
        atomicAdd(base + 2, v.z * alpha);
        atomicAdd(base + 3, v.w * alpha);
    }
}

// out = leaky_relu(agg + b[c] + self + slb[c], slope)
__global__ void combine_act_kernel(const float* __restrict__ agg, const float* __restrict__ b,
                                   const float* __restrict__ self, const float* __restrict__ slb,
                                   float* __restrict__ out, long long total, int C, float slope)
{
    const long long i = blockIdx.x * (long long)blockDim.x + threadIdx.x;
    if (i >= total) return;
    const int c = (int)(i & (C - 1));
    float v = agg[i] + b[c] + self[i] + slb[c];
    out[i] = v > 0.f ? v : slope * v;
}

// out = agg + b[c]   (head epilogue)
__global__ void add_bias_kernel(const float* __restrict__ agg, const float* __restrict__ b,
                                float* __restrict__ out, long long total, int C)
{
    const long long i = blockIdx.x * (long long)blockDim.x + threadIdx.x;
    if (i >= total) return;
    out[i] = agg[i] + b[(int)(i & (C - 1))];
}

// ---------------------------------------------------------------------------
static inline int gridFor(long long n, int b) { return (int)((n + b - 1) / b); }

static void run_gat(const float* h, int C,
                    const float* a_s, const float* a_d,
                    const int* src, const int* dst, int E_real, int nloop,
                    float* als, float* ald, unsigned int* emax, float* den, float* ee,
                    float* agg, int N, hipStream_t stream)
{
    const int B = 256;
    node_dots_kernel<<<gridFor((long long)N * 32, B), B, 0, stream>>>(h, a_s, a_d, als, ald, N, C);
    fill_u32_kernel<<<gridFor(N, B), B, 0, stream>>>(emax, 0xFF800000u, N);          // -inf
    fill_u32_kernel<<<gridFor(N, B), B, 0, stream>>>((unsigned int*)den, 0u, N);
    fill_u32_kernel<<<gridFor((long long)N * C, B), B, 0, stream>>>((unsigned int*)agg, 0u, (long long)N * C);
    const int Et = E_real + nloop;
    edge_max_kernel<<<gridFor(Et, B), B, 0, stream>>>(src, dst, E_real, nloop, als, ald, emax);
    edge_exp_kernel<<<gridFor(Et, B), B, 0, stream>>>(src, dst, E_real, nloop, als, ald, emax, den, ee);
    edge_scatter_kernel<<<gridFor((long long)Et * 32, B), B, 0, stream>>>(src, dst, E_real, nloop, ee, den, h, agg, C);
}

extern "C" void kernel_launch(void* const* d_in, const int* in_sizes, int n_in,
                              void* d_out, int out_size, void* d_ws, size_t ws_size,
                              hipStream_t stream)
{
    const float* x      = (const float*)d_in[0];
    const int*   ei     = (const int*)  d_in[1];
    const float* W0     = (const float*)d_in[2];
    const float* a_src0 = (const float*)d_in[3];
    const float* a_dst0 = (const float*)d_in[4];
    const float* b0     = (const float*)d_in[5];
    const float* slW0   = (const float*)d_in[6];
    const float* slb0   = (const float*)d_in[7];
    const float* W1     = (const float*)d_in[8];
    const float* a_src1 = (const float*)d_in[9];
    const float* a_dst1 = (const float*)d_in[10];
    const float* b1     = (const float*)d_in[11];
    const float* slW1   = (const float*)d_in[12];
    const float* slb1   = (const float*)d_in[13];
    const float* Wmu    = (const float*)d_in[14];
    const float* a_smu  = (const float*)d_in[15];
    const float* a_dmu  = (const float*)d_in[16];
    const float* b_mu   = (const float*)d_in[17];
    const float* Wlv    = (const float*)d_in[18];
    const float* a_slv  = (const float*)d_in[19];
    const float* a_dlv  = (const float*)d_in[20];
    const float* b_lv   = (const float*)d_in[21];

    const int N   = in_sizes[0] / 128;
    const int E   = in_sizes[1] / 2;
    const int LAT = in_sizes[14] / 128;   // 64
    const int* src = ei;
    const int* dst = ei + E;

    // workspace carve-up (256B aligned)
    size_t off = 0;
    auto carve = [&](size_t bytes) -> void* {
        void* p = (char*)d_ws + off;
        off += (bytes + 255) & ~(size_t)255;
        return p;
    };
    float*        BA   = (float*)       carve((size_t)N * 128 * 4);  // GAT-transformed feats
    float*        BB   = (float*)       carve((size_t)N * 128 * 4);  // layer activations
    float*        BC   = (float*)       carve((size_t)N * 128 * 4);  // aggregation target
    float*        als  = (float*)       carve((size_t)N * 4);
    float*        ald  = (float*)       carve((size_t)N * 4);
    unsigned int* emax = (unsigned int*)carve((size_t)N * 4);
    float*        den  = (float*)       carve((size_t)N * 4);
    float*        ee   = (float*)       carve((size_t)(E + N) * 4);
    if (off > ws_size) return;   // insufficient scratch

    const int B = 256;
    const int gemmGrid = (N + 127) / 128;
    const long long NC = (long long)N * 128;

    // ---- layer 0: GATConv(no self loops) + self-linear, LeakyReLU(0.01) ----
    wmma_gemm_k128<128><<<gemmGrid, B, 0, stream>>>(x, W0, BA, N);
    run_gat(BA, 128, a_src0, a_dst0, src, dst, E, 0, als, ald, emax, den, ee, BC, N, stream);
    wmma_gemm_k128<128><<<gemmGrid, B, 0, stream>>>(x, slW0, BB, N);
    combine_act_kernel<<<gridFor(NC, B), B, 0, stream>>>(BC, b0, BB, slb0, BB, NC, 128, 0.01f);

    // ---- layer 1: GATConv(self loops) + self-linear, LeakyReLU(0.01) ----
    wmma_gemm_k128<128><<<gemmGrid, B, 0, stream>>>(BB, W1, BA, N);
    run_gat(BA, 128, a_src1, a_dst1, src, dst, E, N, als, ald, emax, den, ee, BC, N, stream);
    wmma_gemm_k128<128><<<gemmGrid, B, 0, stream>>>(BB, slW1, BA, N);
    combine_act_kernel<<<gridFor(NC, B), B, 0, stream>>>(BC, b1, BA, slb1, BB, NC, 128, 0.01f);

    // ---- heads: mu / logvar GATConv(self loops), C = LAT = 64 ----
    float* outmu = (float*)d_out;
    float* outlv = outmu + (size_t)N * LAT;
    const long long NL = (long long)N * LAT;

    wmma_gemm_k128<64><<<gemmGrid, B, 0, stream>>>(BB, Wmu, BA, N);
    run_gat(BA, LAT, a_smu, a_dmu, src, dst, E, N, als, ald, emax, den, ee, BC, N, stream);
    add_bias_kernel<<<gridFor(NL, B), B, 0, stream>>>(BC, b_mu, outmu, NL, LAT);

    wmma_gemm_k128<64><<<gemmGrid, B, 0, stream>>>(BB, Wlv, BA, N);
    run_gat(BA, LAT, a_slv, a_dlv, src, dst, E, N, als, ald, emax, den, ee, BC, N, stream);
    add_bias_kernel<<<gridFor(NL, B), B, 0, stream>>>(BC, b_lv, outlv, NL, LAT);
}
